// Attention_49005576847990
// MI455X (gfx1250) — matrix operand order repaired and checked by reading, not compiled, not run on hardware
//
#include <hip/hip_runtime.h>
#include <hip/hip_bf16.h>
#include <math.h>

// B=32, L=512, I=1024, H=1024.  All matmuls via v_wmma_f32_16x16x32_bf16.

typedef __bf16 bf16;
typedef __attribute__((ext_vector_type(16))) __bf16 bf16x16;
typedef __attribute__((ext_vector_type(8)))  __bf16 bf16x8;
typedef __attribute__((ext_vector_type(8)))  float  v8f;
typedef __attribute__((ext_vector_type(4)))  float  v4f;

#define DEV static __device__ __forceinline__

#define WMMA_BF16(a, b, c) \
  __builtin_amdgcn_wmma_f32_16x16x32_bf16(false, (a), false, (b), (short)0, (c), false, false)

// ---- fragment loaders (wave32; A/B K order is contiguous, k = 16*(lane/16) + s) -------------

// A 16x32 bf16, row-major source, lane = M, contiguous K {kb..kb+15}, kb = 16*(lane/16)
DEV bf16x16 load_a_bf16(const bf16* A, int ld, int m0, int k0, int lane) {
  int row = m0 + (lane & 15);
  int kb  = (lane & 16) ? 16 : 0;
  const bf16* p = A + (size_t)row * ld + k0 + kb;
  bf16x8 lo = *(const bf16x8*)p;
  bf16x8 hi = *(const bf16x8*)(p + 8);
  bf16x16 r;
#pragma unroll
  for (int j = 0; j < 8; ++j) { r[j] = lo[j]; r[j + 8] = hi[j]; }
  return r;
}

// Same but converting from f32 source on the fly.
DEV bf16x16 load_a_f32(const float* A, int ld, int m0, int k0, int lane) {
  int row = m0 + (lane & 15);
  int kb  = (lane & 16) ? 16 : 0;
  const float* p = A + (size_t)row * ld + k0 + kb;
  v4f a0 = *(const v4f*)p;
  v4f a1 = *(const v4f*)(p + 4);
  v4f b0 = *(const v4f*)(p + 8);
  v4f b1 = *(const v4f*)(p + 12);
  bf16x16 r;
#pragma unroll
  for (int j = 0; j < 4; ++j) {
    r[j]      = (bf16)a0[j];
    r[j + 4]  = (bf16)a1[j];
    r[j + 8]  = (bf16)b0[j];
    r[j + 12] = (bf16)b1[j];
  }
  return r;
}

// B 32x16 where B[k][n] = W[n][k] (dot-of-rows): contiguous 16 K per lane.
DEV bf16x16 load_b_rows(const bf16* W, int ldw, int n0, int k0, int lane) {
  int n  = n0 + (lane & 15);
  int kb = (lane & 16) ? 16 : 0;
  return *(const bf16x16*)(W + (size_t)n * ldw + k0 + kb);
}

// B 32x16 where B[k][n] = V[k][n] (normal second operand): strided per lane.
DEV bf16x16 load_b_cols(const bf16* V, int ldv, int n0, int k0, int lane) {
  int n  = n0 + (lane & 15);
  int kb = (lane & 16) ? 16 : 0;
  const bf16* p = V + (size_t)(k0 + kb) * ldv + n;
  bf16x16 r;
#pragma unroll
  for (int j = 0; j < 16; ++j) r[j] = p[(size_t)j * ldv];
  return r;
}

// D 16x16 f32: lane = N, VGPR r -> row m0 + (lane>=16 ? 8 : 0) + r
DEV void store_d_f32(float* C, int ldc, int m0, int n0, int lane, v8f acc) {
  int col = n0 + (lane & 15);
  int rb  = m0 + ((lane & 16) ? 8 : 0);
#pragma unroll
  for (int r = 0; r < 8; ++r) C[(size_t)(rb + r) * ldc + col] = acc[r];
}

DEV void store_d_bf16(bf16* C, int ldc, int m0, int n0, int lane, v8f acc) {
  int col = n0 + (lane & 15);
  int rb  = m0 + ((lane & 16) ? 8 : 0);
#pragma unroll
  for (int r = 0; r < 8; ++r) C[(size_t)(rb + r) * ldc + col] = (bf16)acc[r];
}

DEV v8f v8f_zero() { v8f z = {0.f,0.f,0.f,0.f,0.f,0.f,0.f,0.f}; return z; }

// ---- small helper kernels ------------------------------------------------

__global__ __launch_bounds__(256) void k_cvt(const float* __restrict__ s,
                                             bf16* __restrict__ d, int n) {
  int i = blockIdx.x * 256 + threadIdx.x;
  if (i < n) d[i] = (bf16)s[i];
}

__global__ __launch_bounds__(256) void k_bias(const float* __restrict__ a,
                                              const float* __restrict__ b,
                                              float* __restrict__ o, int n) {
  int i = blockIdx.x * 256 + threadIdx.x;
  if (i < n) o[i] = a[i] + b[i];
}

// ---- x_proj = inputs @ w_ih^T + bias : [16384 x 4096] bf16 --------------
// grid 8192 x 256 threads = 65536 waves; wave = one 16x64 tile, K=1024.
__global__ __launch_bounds__(256) void k_xproj(const float* __restrict__ A,
                                               const bf16* __restrict__ W,
                                               const float* __restrict__ bias,
                                               bf16* __restrict__ out) {
  int wid  = blockIdx.x * 8 + (threadIdx.x >> 5);
  int lane = threadIdx.x & 31;
  int m0 = (wid >> 6) * 16;   // 1024 row tiles
  int n0 = (wid & 63) * 64;   // 64 col groups
  v8f acc[4];
#pragma unroll
  for (int t = 0; t < 4; ++t) {
    float bv = bias[n0 + t * 16 + (lane & 15)];
#pragma unroll
    for (int r = 0; r < 8; ++r) acc[t][r] = bv;
  }
  for (int k0 = 0; k0 < 1024; k0 += 32) {
    bf16x16 a = load_a_f32(A, 1024, m0, k0, lane);
#pragma unroll
    for (int t = 0; t < 4; ++t) {
      bf16x16 b = load_b_rows(W, 1024, n0 + t * 16, k0, lane);
      acc[t] = WMMA_BF16(a, b, acc[t]);
    }
  }
#pragma unroll
  for (int t = 0; t < 4; ++t) store_d_bf16(out, 4096, m0, n0 + t * 16, lane, acc[t]);
}

// ---- one LSTM timestep (fused gates GEMM + nonlinearity) ----------------
// grid 64 blocks x 256 threads (8 waves). Block owns 16 hidden cols across
// all 4 gates; wave w: gate = w>>1, row-tile = w&1. K loop over H=1024.
__global__ __launch_bounds__(256) void k_lstm_step(
    const bf16* __restrict__ h_in, const bf16* __restrict__ Whh,
    const bf16* __restrict__ xproj, float* __restrict__ c_state,
    bf16* __restrict__ h_out, bf16* __restrict__ dec_bf,
    float* __restrict__ out, int t) {
  __shared__ float gl[32][64];  // [batch][gate*16 + col]
  int lane  = threadIdx.x & 31;
  int wave  = threadIdx.x >> 5;
  int gate  = wave >> 1;
  int m0    = (wave & 1) * 16;
  int hbase = blockIdx.x * 16;
  int n0    = gate * 1024 + hbase;  // column in 4H

  v8f acc = v8f_zero();
  for (int k0 = 0; k0 < 1024; k0 += 32) {
    bf16x16 a = load_a_bf16(h_in, 1024, m0, k0, lane);
    bf16x16 b = load_b_rows(Whh, 1024, n0, k0, lane);
    acc = WMMA_BF16(a, b, acc);
  }
  {
    int col4 = n0 + (lane & 15);
    int lcol = gate * 16 + (lane & 15);
    int rb   = m0 + ((lane & 16) ? 8 : 0);
#pragma unroll
    for (int r = 0; r < 8; ++r) {
      int bidx = rb + r;  // batch index
      float v = acc[r] + (float)xproj[((size_t)bidx * 512 + t) * 4096 + col4];
      gl[bidx][lcol] = v;
    }
  }
  __syncthreads();
#pragma unroll
  for (int e = threadIdx.x; e < 512; e += 256) {
    int row  = e >> 4;        // batch
    int cl   = e & 15;
    int hcol = hbase + cl;
    float gi = gl[row][cl];
    float gf = gl[row][16 + cl];
    float gg = gl[row][32 + cl];
    float go = gl[row][48 + cl];
    float si = 1.0f / (1.0f + __expf(-gi));
    float sf = 1.0f / (1.0f + __expf(-gf));
    float so = 1.0f / (1.0f + __expf(-go));
    float cp = c_state[row * 1024 + hcol];
    float cn = sf * cp + si * tanhf(gg);
    float h  = so * tanhf(cn);
    c_state[row * 1024 + hcol] = cn;
    h_out[row * 1024 + hcol]   = (bf16)h;
    dec_bf[((size_t)row * 512 + t) * 1024 + hcol] = (bf16)h;
    out[((size_t)row * 512 + t) * 2048 + 1024 + hcol] = h;  // dec half of concat
  }
}

// ---- scores[b] = dec[b] @ enc[b]^T : [512 x 512] f32 per batch ----------
__global__ __launch_bounds__(256) void k_scores(const bf16* __restrict__ dec_bf,
                                                const bf16* __restrict__ enc_bf,
                                                float* __restrict__ scores) {
  int wid  = blockIdx.x * 8 + (threadIdx.x >> 5);
  int lane = threadIdx.x & 31;
  int b    = wid >> 8;
  int rem  = wid & 255;
  int m0   = (rem >> 3) * 16;
  int n0   = (rem & 7) * 64;
  const bf16* A  = dec_bf + (size_t)b * 512 * 1024;
  const bf16* Bm = enc_bf + (size_t)b * 512 * 1024;
  v8f acc[4];
#pragma unroll
  for (int t = 0; t < 4; ++t) acc[t] = v8f_zero();
  for (int k0 = 0; k0 < 1024; k0 += 32) {
    bf16x16 a = load_a_bf16(A, 1024, m0, k0, lane);
#pragma unroll
    for (int t = 0; t < 4; ++t) {
      bf16x16 bb = load_b_rows(Bm, 1024, n0 + t * 16, k0, lane);
      acc[t] = WMMA_BF16(a, bb, acc[t]);
    }
  }
  float* C = scores + (size_t)b * 512 * 512;
#pragma unroll
  for (int t = 0; t < 4; ++t) store_d_f32(C, 512, m0, n0 + t * 16, lane, acc[t]);
}

// ---- softmax per row (wave per row), writes P in-place as bf16 ----------
__global__ __launch_bounds__(256) void k_softmax(float* __restrict__ scores) {
  int row  = blockIdx.x * 8 + (threadIdx.x >> 5);
  int lane = threadIdx.x & 31;
  float* base = scores + (size_t)row * 512;
  float v[16];
#pragma unroll
  for (int i = 0; i < 16; ++i) v[i] = base[lane + i * 32];
  float mx = v[0];
#pragma unroll
  for (int i = 1; i < 16; ++i) mx = fmaxf(mx, v[i]);
  for (int off = 16; off > 0; off >>= 1) mx = fmaxf(mx, __shfl_xor(mx, off, 32));
  float s = 0.f;
#pragma unroll
  for (int i = 0; i < 16; ++i) { v[i] = __expf(v[i] - mx); s += v[i]; }
  for (int off = 16; off > 0; off >>= 1) s += __shfl_xor(s, off, 32);
  float inv = 1.0f / s;
  bf16* p = (bf16*)base;  // bf16 P packs into first half of the f32 row
#pragma unroll
  for (int i = 0; i < 16; ++i) p[lane + i * 32] = (bf16)(v[i] * inv);
}

// ---- ctx[b] = P[b] @ enc[b] : [512 x 1024] f32 into first half of out ---
__global__ __launch_bounds__(256) void k_ctx(const bf16* __restrict__ p_bf,
                                             const bf16* __restrict__ enc_bf,
                                             float* __restrict__ out) {
  int wid  = blockIdx.x * 8 + (threadIdx.x >> 5);
  int lane = threadIdx.x & 31;
  int b    = wid >> 9;
  int rem  = wid & 511;
  int m0   = (rem >> 4) * 16;
  int n0   = (rem & 15) * 64;
  const bf16* A = p_bf + (size_t)b * 512 * 1024;   // P row stride = 1024 bf16
  const bf16* V = enc_bf + (size_t)b * 512 * 1024;
  v8f acc[4];
#pragma unroll
  for (int t = 0; t < 4; ++t) acc[t] = v8f_zero();
  for (int k0 = 0; k0 < 512; k0 += 32) {
    bf16x16 a = load_a_bf16(A, 1024, m0, k0, lane);
#pragma unroll
    for (int t = 0; t < 4; ++t) {
      bf16x16 bb = load_b_cols(V, 1024, n0 + t * 16, k0, lane);
      acc[t] = WMMA_BF16(a, bb, acc[t]);
    }
  }
  float* C = out + (size_t)b * 512 * 2048;
#pragma unroll
  for (int t = 0; t < 4; ++t) store_d_f32(C, 2048, m0, n0 + t * 16, lane, acc[t]);
}

// ---- host orchestration --------------------------------------------------

extern "C" void kernel_launch(void* const* d_in, const int* in_sizes, int n_in,
                              void* d_out, int out_size, void* d_ws, size_t ws_size,
                              hipStream_t stream) {
  const float* inputs = (const float*)d_in[0];  // [32,512,1024]
  const float* h0     = (const float*)d_in[1];  // [1,32,1024]
  const float* c0     = (const float*)d_in[2];  // [1,32,1024]
  const float* enc    = (const float*)d_in[3];  // [32,512,1024]
  const float* w_ih   = (const float*)d_in[4];  // [4096,1024]
  const float* w_hh   = (const float*)d_in[5];  // [4096,1024]
  const float* b_ih   = (const float*)d_in[6];  // [4096]
  const float* b_hh   = (const float*)d_in[7];  // [4096]
  float* out = (float*)d_out;                   // [32,512,2048]
  char*  ws  = (char*)d_ws;

  bf16*  w_ih_bf = (bf16*)(ws + 0ull);            //  8 MB
  bf16*  w_hh_bf = (bf16*)(ws + 8388608ull);      //  8 MB
  bf16*  enc_bf  = (bf16*)(ws + 16777216ull);     // 32 MB
  bf16*  dec_bf  = (bf16*)(ws + 50331648ull);     // 32 MB
  bf16*  hbuf0   = (bf16*)(ws + 83886080ull);     // 64 KB
  bf16*  hbuf1   = (bf16*)(ws + 83951616ull);     // 64 KB
  float* c_state = (float*)(ws + 84017152ull);    // 128 KB
  float* bias    = (float*)(ws + 84148224ull);    // 16 KB
  bf16*  xproj   = (bf16*)(ws + 84164608ull);     // 128 MB
  float* scores  = (float*)(ws + 84164608ull);    // aliases xproj (dead after LSTM)

  hipMemcpyAsync(c_state, c0, 32 * 1024 * sizeof(float),
                 hipMemcpyDeviceToDevice, stream);
  k_cvt<<<16384, 256, 0, stream>>>(w_ih, w_ih_bf, 4194304);
  k_cvt<<<16384, 256, 0, stream>>>(w_hh, w_hh_bf, 4194304);
  k_cvt<<<65536, 256, 0, stream>>>(enc, enc_bf, 16777216);
  k_cvt<<<128, 256, 0, stream>>>(h0, hbuf0, 32768);
  k_bias<<<16, 256, 0, stream>>>(b_ih, b_hh, bias, 4096);

  k_xproj<<<8192, 256, 0, stream>>>(inputs, w_ih_bf, bias, xproj);

  for (int t = 0; t < 512; ++t) {
    const bf16* hin = (t & 1) ? hbuf1 : hbuf0;
    bf16* hout      = (t & 1) ? hbuf0 : hbuf1;
    k_lstm_step<<<64, 256, 0, stream>>>(hin, w_hh_bf, xproj, c_state,
                                        hout, dec_bf, out, t);
  }

  k_scores<<<1024, 256, 0, stream>>>(dec_bf, enc_bf, scores);
  k_softmax<<<2048, 256, 0, stream>>>(scores);
  k_ctx<<<2048, 256, 0, stream>>>((const bf16*)scores, enc_bf, out);
}
